// WindowAttention_45741401702772
// MI455X (gfx1250) — matrix operand order repaired and checked by reading, not compiled, not run on hardware
//
#include <hip/hip_runtime.h>
#include <hip/hip_bf16.h>

typedef _Float16 h16;
typedef __attribute__((ext_vector_type(16))) _Float16 v16h;
typedef __attribute__((ext_vector_type(8)))  _Float16 v8h;
typedef __attribute__((ext_vector_type(4)))  _Float16 v4h;
typedef __attribute__((ext_vector_type(8)))  float    v8f;

#define DEVI __device__ __forceinline__

static constexpr int WH = 7, WW = 7, NTOK = 49, C = 128, NH = 4, HD = 32;
static constexpr int NPAD = 64;
static constexpr float SCALE = 0.17677669529663687f; // 1/sqrt(32)

DEVI v8f vzero8() { v8f z = {0.f,0.f,0.f,0.f,0.f,0.f,0.f,0.f}; return z; }

// A-fragment (16x32 f16): lane = row (L&15); contiguous 16 K-values starting at 16*hi
// p points at row_base + k0 within a row-major [rows][K] buffer.
DEVI v16h load_fragA(const h16* p, int hi) {
  return *(const v16h*)(p + 16 * hi);
}

// B-fragment (32x16 f16) loaded from a row-major B^T buffer [N][K]:
// lane = column N (L&15); contiguous 16 K-values starting at 16*hi.
DEVI v16h load_fragB(const h16* p, int hi) {
  return *(const v16h*)(p + 16 * hi);
}

DEVI v8f wmma_f16(v16h a, v16h b, v8f c) {
  return __builtin_amdgcn_wmma_f32_16x16x32_f16(false, a, false, b, (short)0, c,
                                                false, false);
}

// ---------------- prep kernels ----------------

__global__ void prep_weights_kernel(const float* __restrict__ qkv_w,
                                    const float* __restrict__ proj_w,
                                    h16* __restrict__ wq, h16* __restrict__ wp) {
  int i = blockIdx.x * 256 + threadIdx.x;
  if (i < 3 * C * C) wq[i] = (h16)qkv_w[i];   // [384][128] row-major, f16
  if (i < C * C)     wp[i] = (h16)proj_w[i];  // [128][128] row-major, f16
}

__global__ void prep_bias_kernel(const float* __restrict__ table,
                                 float* __restrict__ b64) {
  int i = blockIdx.x * 256 + threadIdx.x;      // NH*64*64 = 16384
  if (i >= NH * NPAD * NPAD) return;
  int h = i >> 12, rem = i & 4095, r = rem >> 6, c = rem & 63;
  float v;
  if (c >= NTOK) {
    v = -1e30f;                                // mask padded key columns
  } else {
    int rr = r < NTOK ? r : NTOK - 1;
    int yi = rr / WW, xi = rr % WW, yj = c / WW, xj = c % WW;
    int idx = (yi - yj + WH - 1) * (2 * WW - 1) + (xi - xj + WW - 1);
    v = table[idx * NH + h];
  }
  b64[i] = v;
}

// ---------------- fused window-attention kernel ----------------

__global__ __launch_bounds__(64) void winattn_kernel(
    const float* __restrict__ x, const float* __restrict__ qkv_b,
    const float* __restrict__ proj_b, const h16* __restrict__ wq,
    const h16* __restrict__ wp, const float* __restrict__ b64,
    float* __restrict__ out, int nWin) {
  struct alignas(32) WaveMem {
    h16 xs[NPAD * C];     // x (f16, zero-padded rows)
    h16 outb[NPAD * C];   // concat head outputs (attention result)
    h16 qb[NPAD * HD];    // q for current head, [token][d]
    h16 kb[NPAD * HD];    // k for current head, [token][d]  (== B^T of k^T)
    h16 vT[HD * NPAD];    // v^T for current head, [d][token] (== B^T of v)
    h16 sA[NPAD * NPAD];  // softmax probs, [q-token][k-token]
  };
  __shared__ WaveMem sm[2];

  const int lane = threadIdx.x & 31;
  const int wid  = threadIdx.x >> 5;
  const int l15  = lane & 15;
  const int hi   = lane >> 4;
  const int rb   = hi * 8;  // C/D fragment row base

  int win = blockIdx.x * 2 + wid;
  if (win >= nWin) win = nWin - 1;  // duplicate work, uniform barriers
  WaveMem& M = sm[wid];

  // ---- stage 0: load x window, fp32 -> f16 into LDS, pad rows >= 49 ----
  {
    const float4* xw = (const float4*)(x + (size_t)win * NTOK * C);
    for (int row = 0; row < NPAD; ++row) {
      float4 f;
      if (row < NTOK) f = xw[row * 32 + lane];
      else            f = make_float4(0.f, 0.f, 0.f, 0.f);
      v4h hv = {(h16)f.x, (h16)f.y, (h16)f.z, (h16)f.w};
      *(v4h*)(&M.xs[row * C + lane * 4]) = hv;
    }
  }
  __syncthreads();

  for (int h = 0; h < NH; ++h) {
    // ---- stage 1: q,k,v = x @ W^T + b for this head ----
    for (int m = 0; m < 3; ++m) {          // 0=q 1=k 2=v
      const int nbase = m * C + h * HD;
      for (int nt = 0; nt < 2; ++nt) {
        v8f acc[4] = {vzero8(), vzero8(), vzero8(), vzero8()};
        for (int ks = 0; ks < 4; ++ks) {
          v16h bf = load_fragB(wq + (size_t)(nbase + nt * 16 + l15) * C + ks * 32, hi);
#pragma unroll
          for (int mt = 0; mt < 4; ++mt) {
            v16h af = load_fragA(&M.xs[(mt * 16 + l15) * C + ks * 32], hi);
            acc[mt] = wmma_f16(af, bf, acc[mt]);
          }
        }
        const int col = nt * 16 + l15;
        const float bias = qkv_b[nbase + col];
#pragma unroll
        for (int mt = 0; mt < 4; ++mt) {
          if (m == 2) {  // store v transposed: vT[d][token], rows contiguous
            v8h hv;
#pragma unroll
            for (int r = 0; r < 8; ++r) hv[r] = (h16)(acc[mt][r] + bias);
            *(v8h*)(&M.vT[col * NPAD + mt * 16 + rb]) = hv;
          } else {
            h16* dst = (m == 0) ? M.qb : M.kb;
#pragma unroll
            for (int r = 0; r < 8; ++r)
              dst[(mt * 16 + rb + r) * HD + col] = (h16)(acc[mt][r] + bias);
          }
        }
      }
    }
    __syncthreads();

    // ---- stage 2: attn = softmax(q k^T * scale + bias) ----
    const float* bh = b64 + h * NPAD * NPAD;
    for (int mt = 0; mt < 4; ++mt) {
      v16h aq = load_fragA(&M.qb[(mt * 16 + l15) * HD], hi);
      v8f at[4];
#pragma unroll
      for (int nt = 0; nt < 4; ++nt) {
        v16h bk = load_fragB(&M.kb[(nt * 16 + l15) * HD], hi);
        at[nt] = wmma_f16(aq, bk, vzero8());
      }
#pragma unroll
      for (int r = 0; r < 8; ++r) {
        const int row = mt * 16 + rb + r;
        float l0 = at[0][r] * SCALE + bh[row * NPAD +  0 + l15];
        float l1 = at[1][r] * SCALE + bh[row * NPAD + 16 + l15];
        float l2 = at[2][r] * SCALE + bh[row * NPAD + 32 + l15];
        float l3 = at[3][r] * SCALE + bh[row * NPAD + 48 + l15];
        float mx = fmaxf(fmaxf(l0, l1), fmaxf(l2, l3));
        mx = fmaxf(mx, __shfl_xor(mx, 1));
        mx = fmaxf(mx, __shfl_xor(mx, 2));
        mx = fmaxf(mx, __shfl_xor(mx, 4));
        mx = fmaxf(mx, __shfl_xor(mx, 8));
        float p0 = __expf(l0 - mx), p1 = __expf(l1 - mx);
        float p2 = __expf(l2 - mx), p3 = __expf(l3 - mx);
        float s = p0 + p1 + p2 + p3;
        s += __shfl_xor(s, 1);
        s += __shfl_xor(s, 2);
        s += __shfl_xor(s, 4);
        s += __shfl_xor(s, 8);
        float inv = 1.0f / s;
        M.sA[row * NPAD +  0 + l15] = (h16)(p0 * inv);
        M.sA[row * NPAD + 16 + l15] = (h16)(p1 * inv);
        M.sA[row * NPAD + 32 + l15] = (h16)(p2 * inv);
        M.sA[row * NPAD + 48 + l15] = (h16)(p3 * inv);
      }
    }

    // ---- stage 3: out_h = attn @ v -> outb[:, h*32 : h*32+32] ----
    for (int mt = 0; mt < 4; ++mt) {
#pragma unroll
      for (int nt2 = 0; nt2 < 2; ++nt2) {
        v8f o = vzero8();
#pragma unroll
        for (int ks2 = 0; ks2 < 2; ++ks2) {
          v16h aa = load_fragA(&M.sA[(mt * 16 + l15) * NPAD + ks2 * 32], hi);
          v16h bv = load_fragB(&M.vT[(nt2 * 16 + l15) * NPAD + ks2 * 32], hi);
          o = wmma_f16(aa, bv, o);
        }
        const int col = h * HD + nt2 * 16 + l15;
#pragma unroll
        for (int r = 0; r < 8; ++r)
          M.outb[(mt * 16 + rb + r) * C + col] = (h16)o[r];
      }
    }
    __syncthreads();
  }

  // ---- stage 4: final projection: out = outb @ proj_w^T + proj_b ----
  float* ow = out + (size_t)win * NTOK * C;
  for (int nt = 0; nt < 8; ++nt) {
    v8f acc[4] = {vzero8(), vzero8(), vzero8(), vzero8()};
    for (int ks = 0; ks < 4; ++ks) {
      v16h bf = load_fragB(wp + (size_t)(nt * 16 + l15) * C + ks * 32, hi);
#pragma unroll
      for (int mt = 0; mt < 4; ++mt) {
        v16h af = load_fragA(&M.outb[(mt * 16 + l15) * C + ks * 32], hi);
        acc[mt] = wmma_f16(af, bf, acc[mt]);
      }
    }
    const float pb = proj_b[nt * 16 + l15];
#pragma unroll
    for (int mt = 0; mt < 4; ++mt) {
#pragma unroll
      for (int r = 0; r < 8; ++r) {
        const int row = mt * 16 + rb + r;
        if (row < NTOK) ow[row * C + nt * 16 + l15] = acc[mt][r] + pb;
      }
    }
  }
}

// ---------------- launcher ----------------

extern "C" void kernel_launch(void* const* d_in, const int* in_sizes, int n_in,
                              void* d_out, int out_size, void* d_ws, size_t ws_size,
                              hipStream_t stream) {
  const float* x          = (const float*)d_in[0];
  const float* qkv_w      = (const float*)d_in[1];
  const float* qkv_b      = (const float*)d_in[2];
  const float* proj_w     = (const float*)d_in[3];
  const float* proj_b     = (const float*)d_in[4];
  const float* bias_table = (const float*)d_in[5];

  // workspace layout (all 32B aligned): qkv_w f16 | proj_w f16 | bias64 f32
  h16*   wq  = (h16*)d_ws;                             // 384*128*2   = 98304 B
  h16*   wp  = (h16*)((char*)d_ws + 98304);            // 128*128*2   = 32768 B
  float* b64 = (float*)((char*)d_ws + 131072);         // 4*64*64*4   = 65536 B

  const int B_ = in_sizes[0] / (NTOK * C);

  prep_weights_kernel<<<192, 256, 0, stream>>>(qkv_w, proj_w, wq, wp);
  prep_bias_kernel<<<64, 256, 0, stream>>>(bias_table, b64);

  winattn_kernel<<<(B_ + 1) / 2, 64, 0, stream>>>(x, qkv_b, proj_b, wq, wp, b64,
                                                  (float*)d_out, B_);
}
